// LIFCell_87522843560962
// MI455X (gfx1250) — matrix operand order repaired and checked by reading, not compiled, not run on hardware
//
#include <hip/hip_runtime.h>
#include <stdint.h>
#include <stddef.h>
#include <type_traits>

typedef __attribute__((ext_vector_type(16))) __bf16 v16bf;
typedef __attribute__((ext_vector_type(8)))  __bf16 v8bf;
typedef __attribute__((ext_vector_type(8)))  float  v8f;
typedef __attribute__((ext_vector_type(4)))  float  v4f;

constexpr int BDIM = 4096;     // batch
constexpr int KDIM = 4096;     // D_IN == N_REC (both K segments are 4096)
constexpr int NDIM = 4096;     // N_REC
constexpr int BM = 128, BN = 128, BK = 32;
constexpr int KT_SEG = KDIM / BK;          // 128 tiles per GEMM segment
constexpr int KTILES = KT_SEG * 2;         // 256: inputs@w_in then z0@w_rec
constexpr float DECAY = 0.95122942450071400910f;   // exp(-DT/TAU)
constexpr float OMD   = 1.0f - DECAY;
constexpr float THR   = 1.0f;
constexpr int   N_REF = 5;
constexpr int   LDK   = BK + 8;            // padded LDS row: 80B = 20 banks

struct SharedTiles {
  __bf16 Ah[BM][LDK];   // A tile, hi bf16,  [m][k]
  __bf16 Al[BM][LDK];   // A tile, lo bf16,  [m][k]  (unused in z0 segment)
  __bf16 Bh[BN][LDK];   // B tile, hi bf16,  transposed [n][k]
  __bf16 Bl[BN][LDK];   // B tile, lo bf16,  transposed [n][k]
};

static __device__ __forceinline__ v16bf cat16(v8bf a, v8bf b) {
  return __builtin_shufflevector(a, b, 0,1,2,3,4,5,6,7,8,9,10,11,12,13,14,15);
}

__global__ __launch_bounds__(256)
void lif_fused_wmma(const float* __restrict__ inputs,
                    const float* __restrict__ v0,
                    const int*   __restrict__ r0,
                    const float* __restrict__ z0,
                    const float* __restrict__ w_in,
                    const float* __restrict__ w_rec,
                    float* __restrict__ out) {
  __shared__ SharedTiles sm;

  const int tid  = threadIdx.x;
  const int lane = tid & 31;
  const int wave = tid >> 5;
  const int wm   = wave & 3;              // 4 waves along M
  const int wn   = wave >> 2;             // 2 waves along N
  const int bn0  = blockIdx.x * BN;
  const int bm0  = blockIdx.y * BM;

  v8f acc[2][4] = {};                     // 32x64 wave tile, fp32 accum
  v4f stA[4], stB[4];                     // register-staged next tile

  auto global_stage = [&](int kt) {
    const float* Aseg = (kt < KT_SEG) ? inputs : z0;
    const float* Wseg = (kt < KT_SEG) ? w_in   : w_rec;
    const int kb = (kt & (KT_SEG - 1)) * BK;
    #pragma unroll
    for (int i = 0; i < 4; ++i) {
      const int c  = tid + i * 256;
      const int am = c >> 3, ak = (c & 7) << 2;       // A: 128 rows x 32 k
      stA[i] = *(const v4f*)(Aseg + (size_t)(bm0 + am) * KDIM + kb + ak);
      const int bk = c >> 5, bn = (c & 31) << 2;      // B: 32 k-rows x 128 n
      stB[i] = *(const v4f*)(Wseg + (size_t)(kb + bk) * NDIM + bn0 + bn);
    }
    if (kt + 2 < KTILES) {                // L2 prefetch two tiles ahead
      const int kt2 = kt + 2;
      const float* A2 = (kt2 < KT_SEG) ? inputs : z0;
      const float* W2 = (kt2 < KT_SEG) ? w_in   : w_rec;
      const int kb2 = (kt2 & (KT_SEG - 1)) * BK;
      __builtin_prefetch(A2 + (size_t)(bm0 + (tid >> 3)) * KDIM + kb2 + ((tid & 7) << 2), 0, 1);
      __builtin_prefetch(W2 + (size_t)(kb2 + (tid >> 5)) * NDIM + bn0 + ((tid & 31) << 2), 0, 1);
    }
  };

  // One K-tile step. HAS_LO is a compile-time tag: segment 1 (dense fp32 A)
  // runs the 3-term split-bf16 product; segment 2 (binary z0) drops every
  // A-lo conversion/store/load/WMMA, keeping the inner WMMA chain branch-free.
  auto step = [&](auto has_lo_tag, int kt) {
    constexpr bool HAS_LO = decltype(has_lo_tag)::value;

    // ---- commit register-staged tile (split to bf16 hi/lo) into LDS
    #pragma unroll
    for (int i = 0; i < 4; ++i) {
      const int c  = tid + i * 256;
      const int am = c >> 3, ak = (c & 7) << 2;
      #pragma unroll
      for (int j = 0; j < 4; ++j) {
        const float f = stA[i][j];
        const __bf16 h = (__bf16)f;
        sm.Ah[am][ak + j] = h;
        if constexpr (HAS_LO) sm.Al[am][ak + j] = (__bf16)(f - (float)h);
      }
      const int bk = c >> 5, bn = (c & 31) << 2;
      #pragma unroll
      for (int j = 0; j < 4; ++j) {
        const float f = stB[i][j];
        const __bf16 h = (__bf16)f;
        sm.Bh[bn + j][bk] = h;                 // transpose into [n][k]
        sm.Bl[bn + j][bk] = (__bf16)(f - (float)h);
      }
    }
    __syncthreads();
    if (kt + 1 < KTILES) global_stage(kt + 1);   // overlap VMEM with WMMA

    // ---- A fragments: lane holds row M=lane&15; K runs {0..7,16..23} / {8..15,24..31}
    const int akb = (lane < 16) ? 0 : 8;
    const int am0 = wm * 32 + (lane & 15);
    v16bf afh[2], afl[2];
    #pragma unroll
    for (int mi = 0; mi < 2; ++mi) {
      const __bf16* ph = &sm.Ah[am0 + mi * 16][akb];
      afh[mi] = cat16(*(const v8bf*)ph, *(const v8bf*)(ph + 16));
      if constexpr (HAS_LO) {
        const __bf16* pl = &sm.Al[am0 + mi * 16][akb];
        afl[mi] = cat16(*(const v8bf*)pl, *(const v8bf*)(pl + 16));
      }
    }
    // ---- B fragments: lane holds col N=lane&15; K runs {0..7,16..23} / {8..15,24..31}
    const int bkb = (lane < 16) ? 0 : 8;
    const int bnl = wn * 64 + (lane & 15);
    v16bf bfh[4], bfl[4];
    #pragma unroll
    for (int ni = 0; ni < 4; ++ni) {
      const __bf16* ph = &sm.Bh[bnl + ni * 16][bkb];
      const __bf16* pl = &sm.Bl[bnl + ni * 16][bkb];
      bfh[ni] = cat16(*(const v8bf*)ph, *(const v8bf*)(ph + 16));
      bfl[ni] = cat16(*(const v8bf*)pl, *(const v8bf*)(pl + 16));
    }

    // ---- branch-free WMMA chains
    #pragma unroll
    for (int mi = 0; mi < 2; ++mi) {
      #pragma unroll
      for (int ni = 0; ni < 4; ++ni) {
        acc[mi][ni] = __builtin_amdgcn_wmma_f32_16x16x32_bf16(
            false, afh[mi], false, bfh[ni], (short)0, acc[mi][ni], false, false);
        acc[mi][ni] = __builtin_amdgcn_wmma_f32_16x16x32_bf16(
            false, afh[mi], false, bfl[ni], (short)0, acc[mi][ni], false, false);
        if constexpr (HAS_LO)
          acc[mi][ni] = __builtin_amdgcn_wmma_f32_16x16x32_bf16(
              false, afl[mi], false, bfh[ni], (short)0, acc[mi][ni], false, false);
      }
    }
    __syncthreads();
  };

  global_stage(0);
  for (int kt = 0; kt < KT_SEG; ++kt)
    step(std::integral_constant<bool, true>{}, kt);     // inputs @ w_in
  for (int kt = KT_SEG; kt < KTILES; ++kt)
    step(std::integral_constant<bool, false>{}, kt);    // z0 @ w_rec (A exact in bf16)

  // ---- Fused LIF epilogue ----------------------------------------------
  float* out_v = out;
  float* out_z = out +     (size_t)BDIM * NDIM;
  float* out_r = out + 2 * (size_t)BDIM * NDIM;
  const int mhalf = (lane >> 4) << 3;     // lanes 16..31 hold M+8

  #pragma unroll
  for (int mi = 0; mi < 2; ++mi) {
    #pragma unroll
    for (int ni = 0; ni < 4; ++ni) {
      const int gcol = bn0 + wn * 64 + ni * 16 + (lane & 15);
      const float wdiag = w_rec[(size_t)gcol * NDIM + gcol];   // diagonal disconnect
      #pragma unroll
      for (int e = 0; e < 8; ++e) {
        const int grow = bm0 + wm * 32 + mi * 16 + e + mhalf;
        const size_t idx = (size_t)grow * NDIM + gcol;
        const float s  = acc[mi][ni][e];
        const float zp = z0[idx];
        const float vp = v0[idx];
        const int   rp = r0[idx];
        const float vnew = DECAY * vp + OMD * (s - zp * wdiag) - THR * zp;
        float znew = ((vnew - THR) > 0.0f) ? 1.0f : 0.0f;   // THR==1: /THR is no-op
        if (rp > 0) znew = 0.0f;
        int rnew = rp - 1 + ((int)znew) * N_REF;
        rnew = rnew < 0 ? 0 : (rnew > N_REF ? N_REF : rnew);
        out_v[idx] = vnew;
        out_z[idx] = znew;
        out_r[idx] = (float)rnew;
      }
    }
  }
}

extern "C" void kernel_launch(void* const* d_in, const int* in_sizes, int n_in,
                              void* d_out, int out_size, void* d_ws, size_t ws_size,
                              hipStream_t stream) {
  const float* inputs = (const float*)d_in[0];
  const float* v0     = (const float*)d_in[1];
  const int*   r0     = (const int*)  d_in[2];
  const float* z0     = (const float*)d_in[3];
  const float* w_in   = (const float*)d_in[4];
  const float* w_rec  = (const float*)d_in[5];
  float* out = (float*)d_out;

  dim3 grid(NDIM / BN, BDIM / BM);   // 32 x 32 blocks
  lif_fused_wmma<<<grid, 256, 0, stream>>>(inputs, v0, r0, z0, w_in, w_rec, out);
}
